// PointNet2_29532195127600
// MI455X (gfx1250) — compile-verified, matrix operand order checked by reading, not run on hardware
//
#include <hip/hip_runtime.h>

// ---------------------------------------------------------------------------
// PointNet++ MSG segmentation forward, CDNA5 (gfx1250), wave32 + WMMA.
//
// GEMM strategy (MI455X reasoning): operands are pre-packed to zero-padded
// f16 (Cip = ceil32(Ci), Cop = ceil16(Co)) so the WMMA inner loop is pure
// data movement: per lane, each 16x16x32 fragment = two contiguous 16-byte
// chunks (ISA 7.12.2 layout), loaded as global_load_b128 with no masking and
// no conversion. Each wave computes a 32x16 macro-tile (2 WMMAs share one B
// fragment) to double arithmetic intensity. f32 accumulate; BN re-normalizes
// each layer, so f16 operands are numerically safe. All reductions are
// fixed-order (deterministic). All scratch lives in d_ws.
// ---------------------------------------------------------------------------

typedef __attribute__((ext_vector_type(16))) _Float16 v16h;
typedef __attribute__((ext_vector_type(8)))  _Float16 v8h;
typedef __attribute__((ext_vector_type(8)))  float    v8f;

#define BN_EPS 1e-5f

// --------------------------- transpose (B,3,N)->(B,N,3) --------------------
__global__ void transpose_xyz_kernel(const float* __restrict__ in,
                                     float* __restrict__ out,
                                     int N, int total /* B*N*3 */) {
  int t = blockIdx.x * blockDim.x + threadIdx.x;
  if (t >= total) return;
  int c = t % 3;
  int n = (t / 3) % N;
  int b = t / (3 * N);
  out[t] = in[((size_t)b * 3 + c) * N + n];
}

// --------------------------- farthest point sampling -----------------------
// One block per batch; LDS running min-distance (N <= 4096). Matches the
// jax.lax.scan semantics (emit index BEFORE update; argmax = first max).
__global__ void fps_kernel(const float* __restrict__ xyz /* B,N,3 */,
                           int N, int S, int* __restrict__ out /* B,S */) {
  const int TPB = 256;
  __shared__ float dist[4096];
  __shared__ float rv[TPB];
  __shared__ int   ri[TPB];
  int b = blockIdx.x;
  const float* p = xyz + (size_t)b * N * 3;
  for (int i = threadIdx.x; i < N; i += TPB) dist[i] = 1e10f;
  __syncthreads();
  int far = 0;
  for (int it = 0; it < S; ++it) {
    if (threadIdx.x == 0) out[b * S + it] = far;
    float cx = p[far * 3 + 0], cy = p[far * 3 + 1], cz = p[far * 3 + 2];
    float bestv = -1.0f; int besti = 0;
    for (int i = threadIdx.x; i < N; i += TPB) {
      float dx = p[i * 3 + 0] - cx;
      float dy = p[i * 3 + 1] - cy;
      float dz = p[i * 3 + 2] - cz;
      float d  = dx * dx + dy * dy + dz * dz;
      float nd = fminf(dist[i], d);
      dist[i] = nd;
      if (nd > bestv) { bestv = nd; besti = i; }
    }
    rv[threadIdx.x] = bestv; ri[threadIdx.x] = besti;
    __syncthreads();
    for (int st = TPB / 2; st > 0; st >>= 1) {
      if (threadIdx.x < st) {
        float ov = rv[threadIdx.x + st]; int oi = ri[threadIdx.x + st];
        if (ov > rv[threadIdx.x] ||
            (ov == rv[threadIdx.x] && oi < ri[threadIdx.x])) {
          rv[threadIdx.x] = ov; ri[threadIdx.x] = oi;
        }
      }
      __syncthreads();
    }
    far = ri[0];
    __syncthreads();
  }
}

// --------------------------- gather 3-vectors by index ---------------------
__global__ void gather3_kernel(const float* __restrict__ src /* B,N,3 */,
                               const int* __restrict__ idx /* B,S */,
                               int N, int S, float* __restrict__ out /* B,S,3 */,
                               int total /* B*S*3 */) {
  int t = blockIdx.x * blockDim.x + threadIdx.x;
  if (t >= total) return;
  int c = t % 3;
  int s = (t / 3) % S;
  int b = t / (3 * S);
  int gi = idx[b * S + s];
  out[t] = src[((size_t)b * N + gi) * 3 + c];
}

// --------------------------- ball query ------------------------------------
// First K indices (ascending) with sqdist <= r^2; pad with the first found.
__global__ void ball_query_kernel(const float* __restrict__ xyz /* B,N,3 */,
                                  const float* __restrict__ nxyz /* B,S,3 */,
                                  int N, int S, float r2, int K,
                                  int total /* B*S */, int* __restrict__ gidx) {
  int t = blockIdx.x * blockDim.x + threadIdx.x;
  if (t >= total) return;
  int b = t / S;
  float qx = nxyz[t * 3 + 0], qy = nxyz[t * 3 + 1], qz = nxyz[t * 3 + 2];
  const float* p = xyz + (size_t)b * N * 3;
  int* g = gidx + (size_t)t * K;
  int cnt = 0, first = 0;
  for (int n = 0; n < N && cnt < K; ++n) {
    float dx = p[n * 3 + 0] - qx;
    float dy = p[n * 3 + 1] - qy;
    float dz = p[n * 3 + 2] - qz;
    float d = dx * dx + dy * dy + dz * dz;
    if (d <= r2) { if (cnt == 0) first = n; g[cnt++] = n; }
  }
  for (int k = cnt; k < K; ++k) g[k] = first;
}

// ------------- group: concat(points[gi], xyz[gi]-new_xyz) -> packed f16 ----
__global__ void group_pack_kernel(const float* __restrict__ xyz /* B,N,3 */,
                                  const float* __restrict__ nxyz /* B,S,3 */,
                                  const float* __restrict__ pts /* B,N,Cp */,
                                  const int* __restrict__ gidx /* B,S,K */,
                                  int Cp, int N, int S, int K, int Cip,
                                  int total /* B*S*K */,
                                  _Float16* __restrict__ out /* total x Cip */) {
  int t = blockIdx.x * blockDim.x + threadIdx.x;
  if (t >= total) return;
  int s = (t / K) % S;
  int b = t / (K * S);
  int gi = gidx[t];
  _Float16* o = out + (size_t)t * Cip;
  const float* src = pts + ((size_t)b * N + gi) * Cp;
  for (int c = 0; c < Cp; ++c) o[c] = (_Float16)src[c];
  const float* px = xyz + ((size_t)b * N + gi) * 3;
  const float* qx = nxyz + ((size_t)b * S + s) * 3;
  o[Cp + 0] = (_Float16)(px[0] - qx[0]);
  o[Cp + 1] = (_Float16)(px[1] - qx[1]);
  o[Cp + 2] = (_Float16)(px[2] - qx[2]);
  for (int c = Cp + 3; c < Cip; ++c) o[c] = (_Float16)0.f;
}

// --------------------------- pack weights: f32 CoxCi -> f16 CopxCip --------
__global__ void pack_w_kernel(const float* __restrict__ W, int Co, int Ci,
                              _Float16* __restrict__ out, int Cip,
                              int total /* Cop*Cip */) {
  int t = blockIdx.x * blockDim.x + threadIdx.x;
  if (t >= total) return;
  int c = t % Cip;
  int o = t / Cip;
  float v = (o < Co && c < Ci) ? W[(size_t)o * Ci + c] : 0.f;
  out[t] = (_Float16)v;
}

// --------------------------- WMMA GEMM: Y = A * W^T + bias -----------------
// A: M x Cip f16 (zero padded), Wp: Cop x Cip f16 (zero padded), Y: M x Co.
// One wave per 32x16 macro-tile; fragments are two b128 loads each, per the
// CDNA5 16-bit layout: lane(hi,lr) holds K = {0..7,16..23} + 8*hi.
__global__ void wmma_gemm_f16_kernel(const _Float16* __restrict__ A,
                                     const _Float16* __restrict__ Wp,
                                     const float* __restrict__ bias,
                                     float* __restrict__ Y,
                                     int M, int Cip, int Co, int Cop) {
  const int lane = threadIdx.x & 31;
  const int wave = threadIdx.x >> 5;
  const int mP = (M + 31) >> 5;
  const int oT = Cop >> 4;
  int tile = blockIdx.x * (blockDim.x >> 5) + wave;
  if (tile >= mP * oT) return;                 // wave-uniform exit
  const int mBase = (tile % mP) << 5;
  const int oBase = (tile / mP) << 4;
  const int hi = lane >> 4;
  const int lr = lane & 15;
  const int m0 = mBase + lr;
  const bool has2 = (mBase + 16) < M;          // all M here are 32-multiples
  const int m1 = has2 ? (m0 + 16) : m0;
  const int n  = oBase + lr;
  const _Float16* a0r = A  + (size_t)m0 * Cip + 8 * hi;
  const _Float16* a1r = A  + (size_t)m1 * Cip + 8 * hi;
  const _Float16* wr  = Wp + (size_t)n  * Cip + 8 * hi;
  v8f acc0 = {}, acc1 = {};
  for (int kk = 0; kk < Cip; kk += 32) {
    __builtin_prefetch(a0r + kk + 256, 0, 1);  // global_prefetch_b8
    __builtin_prefetch(wr  + kk + 256, 0, 1);
    union { v16h v; v8h h[2]; } ua0, ua1, ub;
    const v8h* pa0 = (const v8h*)(a0r + kk);
    const v8h* pa1 = (const v8h*)(a1r + kk);
    const v8h* pb  = (const v8h*)(wr  + kk);
    ua0.h[0] = pa0[0]; ua0.h[1] = pa0[2];      // K 0..7 | 16..23 (+8*hi)
    ua1.h[0] = pa1[0]; ua1.h[1] = pa1[2];
    ub.h[0]  = pb[0];  ub.h[1]  = pb[2];
    acc0 = __builtin_amdgcn_wmma_f32_16x16x32_f16(false, ua0.v, false, ub.v,
                                                  (short)0, acc0, false, false);
    acc1 = __builtin_amdgcn_wmma_f32_16x16x32_f16(false, ua1.v, false, ub.v,
                                                  (short)0, acc1, false, false);
  }
  const bool nOk = (n < Co);
  float bb = (bias && nOk) ? bias[n] : 0.f;
#pragma unroll
  for (int r = 0; r < 8; ++r) {                // C/D layout: VGPR r -> M=r(+8)
    int mo = mBase + r + 8 * hi;
    if (nOk)         Y[(size_t)mo * Co + n]        = acc0[r] + bb;
    if (nOk && has2) Y[(size_t)(mo + 16) * Co + n] = acc1[r] + bb;
  }
}

// --------------------------- BN stats (per-channel mean/var) ---------------
__global__ void bn_stats_kernel(const float* __restrict__ Y, int M, int Co,
                                float* __restrict__ mean, float* __restrict__ var) {
  __shared__ float ss[256];
  __shared__ float ss2[256];
  int o = blockIdx.x;
  float s = 0.f, s2 = 0.f;
  for (int r = threadIdx.x; r < M; r += 256) {
    float v = Y[(size_t)r * Co + o];
    s += v; s2 += v * v;
  }
  ss[threadIdx.x] = s; ss2[threadIdx.x] = s2;
  __syncthreads();
  for (int st = 128; st > 0; st >>= 1) {
    if (threadIdx.x < st) {
      ss[threadIdx.x]  += ss[threadIdx.x + st];
      ss2[threadIdx.x] += ss2[threadIdx.x + st];
    }
    __syncthreads();
  }
  if (threadIdx.x == 0) {
    float mu = ss[0] / (float)M;
    mean[o] = mu;
    var[o]  = ss2[0] / (float)M - mu * mu;
  }
}

// ---------- BN+ReLU: update Y (f32, in place) and emit packed f16 ----------
__global__ void bn_apply_pack_kernel(float* __restrict__ Y, int Co, int Cip,
                                     size_t total /* M*Cip */,
                                     const float* __restrict__ mean,
                                     const float* __restrict__ var,
                                     const float* __restrict__ g,
                                     const float* __restrict__ be,
                                     _Float16* __restrict__ pack) {
  size_t t = (size_t)blockIdx.x * blockDim.x + threadIdx.x;
  if (t >= total) return;
  int c = (int)(t % Cip);
  size_t m = t / Cip;
  float v = 0.f;
  if (c < Co) {
    float y  = Y[m * Co + c];
    float xh = (y - mean[c]) * rsqrtf(var[c] + BN_EPS);
    v = xh * g[c] + be[c];
    v = v > 0.f ? v : 0.f;
    Y[m * Co + c] = v;
  }
  pack[t] = (_Float16)v;
}

// --------------------------- max over K neighbors --------------------------
__global__ void maxk_kernel(const float* __restrict__ Y /* (B*S*K) x C */,
                            int S, int K, int C, int total /* B*S*C */,
                            float* __restrict__ out, int Ctot, int coff) {
  int t = blockIdx.x * blockDim.x + threadIdx.x;
  if (t >= total) return;
  int o = t % C;
  int s = (t / C) % S;
  int b = t / (C * S);
  size_t base = ((size_t)(b * S + s) * K) * C + o;
  float m = Y[base];
  for (int k = 1; k < K; ++k) m = fmaxf(m, Y[base + (size_t)k * C]);
  out[((size_t)b * S + s) * Ctot + coff + o] = m;
}

// --------------------------- 3-NN (indices + normalized weights) -----------
__global__ void knn3_kernel(const float* __restrict__ xyz1 /* B,N1,3 */,
                            const float* __restrict__ xyz2 /* B,S2,3 */,
                            int N1, int S2, int total /* B*N1 */,
                            int* __restrict__ oidx, float* __restrict__ ow) {
  int t = blockIdx.x * blockDim.x + threadIdx.x;
  if (t >= total) return;
  int b = t / N1;
  float qx = xyz1[t * 3 + 0], qy = xyz1[t * 3 + 1], qz = xyz1[t * 3 + 2];
  const float* p = xyz2 + (size_t)b * S2 * 3;
  float d0 = 3.4e38f, d1 = 3.4e38f, d2 = 3.4e38f;
  int i0 = 0, i1 = 0, i2 = 0;
  for (int s = 0; s < S2; ++s) {
    float dx = p[s * 3 + 0] - qx;
    float dy = p[s * 3 + 1] - qy;
    float dz = p[s * 3 + 2] - qz;
    float d = dx * dx + dy * dy + dz * dz;
    if (d < d0)      { d2 = d1; i2 = i1; d1 = d0; i1 = i0; d0 = d; i0 = s; }
    else if (d < d1) { d2 = d1; i2 = i1; d1 = d;  i1 = s; }
    else if (d < d2) { d2 = d;  i2 = s; }
  }
  float w0 = 1.f / (d0 + 1e-8f);
  float w1 = 1.f / (d1 + 1e-8f);
  float w2 = 1.f / (d2 + 1e-8f);
  float inv = 1.f / (w0 + w1 + w2);
  oidx[t * 3 + 0] = i0; oidx[t * 3 + 1] = i1; oidx[t * 3 + 2] = i2;
  ow[t * 3 + 0] = w0 * inv; ow[t * 3 + 1] = w1 * inv; ow[t * 3 + 2] = w2 * inv;
}

// ------------------ interpolate + concat -> packed f16 ---------------------
__global__ void interp_concat_pack_kernel(const float* __restrict__ p1,
                                          const float* __restrict__ p2,
                                          const int* __restrict__ idx,
                                          const float* __restrict__ w,
                                          int N1, int S2, int C1, int C2, int Cip,
                                          size_t total /* B*N1*Cip */,
                                          _Float16* __restrict__ out) {
  size_t t = (size_t)blockIdx.x * blockDim.x + threadIdx.x;
  if (t >= total) return;
  int c = (int)(t % Cip);
  int n = (int)((t / Cip) % N1);
  int b = (int)(t / ((size_t)Cip * N1));
  int C = C1 + C2;
  float v = 0.f;
  if (c < C1) {
    v = p1[((size_t)b * N1 + n) * C1 + c];
  } else if (c < C) {
    int c2 = c - C1;
    size_t base = (size_t)b * S2 * C2;
    size_t q = ((size_t)b * N1 + n) * 3;
    v  = w[q + 0] * p2[base + (size_t)idx[q + 0] * C2 + c2];
    v += w[q + 1] * p2[base + (size_t)idx[q + 1] * C2 + c2];
    v += w[q + 2] * p2[base + (size_t)idx[q + 2] * C2 + c2];
  }
  out[t] = (_Float16)v;
}

// ===========================================================================
// Host orchestration
// ===========================================================================
namespace {
struct Lyr { const float *W, *b, *be, *g; int ci, co; };
inline int up16(int x) { return (x + 15) & ~15; }
inline int up32(int x) { return (x + 31) & ~31; }
}

extern "C" void kernel_launch(void* const* d_in, const int* in_sizes, int n_in,
                              void* d_out, int out_size, void* d_ws, size_t ws_size,
                              hipStream_t stream) {
  (void)in_sizes; (void)n_in; (void)out_size; (void)ws_size;
  const int B = 8, N0 = 4096;

  // ---- unpack params (jax tree-flatten order: sorted dict keys) -----------
  int cur = 0;
  auto nf = [&]() { return (const float*)d_in[cur++]; };
  const float* bn1_be  = nf();
  const float* bn1_g   = nf();
  const float* conv1_W = nf();
  const float* conv1_b = nf();
  const float* conv2_W = nf();
  const float* conv2_b = nf();
  auto getL = [&](int ci, int co) {
    Lyr l; l.W = nf(); l.b = nf(); l.be = nf(); l.g = nf(); l.ci = ci; l.co = co;
    return l;
  };
  Lyr fp1[3] = { getL(128, 128), getL(128, 128), getL(128, 128) };
  Lyr fp2[2] = { getL(352, 256), getL(256, 128) };
  Lyr fp3[2] = { getL(512, 256), getL(256, 256) };
  Lyr fp4[2] = { getL(1536, 256), getL(256, 256) };
  Lyr sa1[2][3] = { { getL(6, 16),  getL(16, 16),  getL(16, 32)  },
                    { getL(6, 32),  getL(32, 32),  getL(32, 64)  } };
  Lyr sa2[2][3] = { { getL(99, 64), getL(64, 64),  getL(64, 128) },
                    { getL(99, 64), getL(64, 96),  getL(96, 128) } };
  Lyr sa3[2][3] = { { getL(259, 128), getL(128, 196), getL(196, 256) },
                    { getL(259, 128), getL(128, 196), getL(196, 256) } };
  Lyr sa4[2][3] = { { getL(515, 256), getL(256, 256), getL(256, 512) },
                    { getL(515, 256), getL(256, 384), getL(384, 512) } };
  const float* xyz_in = nf();   // (B, 3, N0)

  // ---- workspace carve-up -------------------------------------------------
  char* base = (char*)d_ws;
  size_t off = 0;
  auto alloc = [&](size_t bytes) -> void* {
    void* p = base + off;
    off = (off + bytes + 255) & ~(size_t)255;
    return p;
  };
  float* x0     = (float*)alloc((size_t)B * N0 * 3 * 4);
  float* l1_xyz = (float*)alloc((size_t)B * 1024 * 3 * 4);
  float* l1p    = (float*)alloc((size_t)B * 1024 * 96 * 4);
  float* l2_xyz = (float*)alloc((size_t)B * 256 * 3 * 4);
  float* l2p    = (float*)alloc((size_t)B * 256 * 256 * 4);
  float* l3_xyz = (float*)alloc((size_t)B * 64 * 3 * 4);
  float* l3p    = (float*)alloc((size_t)B * 64 * 512 * 4);
  float* l4_xyz = (float*)alloc((size_t)B * 16 * 3 * 4);
  float* l4p    = (float*)alloc((size_t)B * 16 * 1024 * 4);
  float* l3f    = (float*)alloc((size_t)B * 64 * 256 * 4);     // fp4 out
  float* l2f    = (float*)alloc((size_t)B * 256 * 256 * 4);    // fp3 out
  float* l1f    = (float*)alloc((size_t)B * 1024 * 128 * 4);   // fp2 out
  float* l0p    = (float*)alloc((size_t)B * N0 * 128 * 4);     // fp1 out
  int*   fps_i  = (int*)alloc((size_t)B * 1024 * 4);
  int*   gidx   = (int*)alloc((size_t)B * 1024 * 32 * 4);
  int*   knn_i  = (int*)alloc((size_t)B * N0 * 3 * 4);
  float* knn_w  = (float*)alloc((size_t)B * N0 * 3 * 4);
  float* meanb  = (float*)alloc(512 * 4);
  float* varb   = (float*)alloc(512 * 4);
  _Float16* wPack = (_Float16*)alloc((size_t)512 * 1024 * 2);  // >= 256*1536 halfs
  const size_t PACK = (size_t)B * 1024 * 32 * 64;              // 16.7M halfs
  _Float16* packA = (_Float16*)alloc(PACK * 2);
  _Float16* packB = (_Float16*)alloc(PACK * 2);
  float* bufY = (float*)alloc(PACK * 4);                       // 16.7M floats

  // ---- helpers ------------------------------------------------------------
  // Runs GEMM (+opt BN+ReLU). inPack has Cip=up32(L.ci); outPack gets
  // Cip=up32(L.co). bufY receives the layer's f32 activation.
  auto run_layer = [&](const _Float16* inPack, const Lyr& L, int M, bool bn,
                       _Float16* outPack, float* Yout) {
    int Cip = up32(L.ci);
    int Cop = up16(L.co);
    { int tot = Cop * Cip;
      pack_w_kernel<<<dim3((tot + 255) / 256), dim3(256), 0, stream>>>(
          L.W, L.co, L.ci, wPack, Cip, tot); }
    int mP = (M + 31) >> 5, oT = Cop >> 4;
    int tiles = mP * oT;
    wmma_gemm_f16_kernel<<<dim3((tiles + 7) / 8), dim3(256), 0, stream>>>(
        inPack, wPack, L.b, Yout, M, Cip, L.co, Cop);
    if (bn) {
      bn_stats_kernel<<<dim3(L.co), dim3(256), 0, stream>>>(Yout, M, L.co, meanb, varb);
      int CipOut = up32(L.co);
      size_t tot = (size_t)M * CipOut;
      bn_apply_pack_kernel<<<dim3((unsigned)((tot + 255) / 256)), dim3(256), 0, stream>>>(
          Yout, L.co, CipOut, tot, meanb, varb, L.g, L.be, outPack);
    }
  };

  auto run_sa = [&](const float* xyzL, int Nin, const float* ptsL, int Cp, int S,
                    const float* rads, const int* Ks, Lyr br[2][3],
                    float* outXyz, float* outPts, int Ctot) {
    fps_kernel<<<dim3(B), dim3(256), 0, stream>>>(xyzL, Nin, S, fps_i);
    int g3 = B * S * 3;
    gather3_kernel<<<dim3((g3 + 255) / 256), dim3(256), 0, stream>>>(
        xyzL, fps_i, Nin, S, outXyz, g3);
    int coff = 0;
    for (int bi = 0; bi < 2; ++bi) {
      int K = Ks[bi];
      float r2 = rads[bi] * rads[bi];
      int q = B * S;
      ball_query_kernel<<<dim3((q + 63) / 64), dim3(64), 0, stream>>>(
          xyzL, outXyz, Nin, S, r2, K, q, gidx);
      int M = B * S * K;
      int Cip0 = up32(Cp + 3);
      group_pack_kernel<<<dim3((M + 127) / 128), dim3(128), 0, stream>>>(
          xyzL, outXyz, ptsL, gidx, Cp, Nin, S, K, Cip0, M, packA);
      const _Float16* in = packA;
      _Float16* nxt = packB;
      for (int l = 0; l < 3; ++l) {
        run_layer(in, br[bi][l], M, true, nxt, bufY);
        const _Float16* tmp = in; in = nxt; nxt = (_Float16*)tmp;
      }
      int Cl = br[bi][2].co;
      int tot = B * S * Cl;
      maxk_kernel<<<dim3((tot + 255) / 256), dim3(256), 0, stream>>>(
          bufY, S, K, Cl, tot, outPts, Ctot, coff);
      coff += Cl;
    }
  };

  // Returns the packed f16 of the final layer output (for chaining the head).
  auto run_fp = [&](const float* xyz1, int N1, const float* xyz2, int S2,
                    const float* p1, int C1, const float* p2, int C2,
                    const Lyr* layers, int nl, float* outPts) -> const _Float16* {
    int q = B * N1;
    knn3_kernel<<<dim3((q + 127) / 128), dim3(128), 0, stream>>>(
        xyz1, xyz2, N1, S2, q, knn_i, knn_w);
    int Cip0 = up32(C1 + C2);
    size_t tot = (size_t)q * Cip0;
    interp_concat_pack_kernel<<<dim3((unsigned)((tot + 255) / 256)), dim3(256), 0, stream>>>(
        p1, p2, knn_i, knn_w, N1, S2, C1, C2, Cip0, tot, packA);
    const _Float16* in = packA;
    _Float16* nxt = packB;
    for (int l = 0; l < nl; ++l) {
      run_layer(in, layers[l], q, true, nxt, bufY);
      const _Float16* tmp = in; in = nxt; nxt = (_Float16*)tmp;
    }
    hipMemcpyAsync(outPts, bufY, (size_t)q * layers[nl - 1].co * 4,
                   hipMemcpyDeviceToDevice, stream);
    return in;   // packed f16 of final activation
  };

  // ---- pipeline -----------------------------------------------------------
  int tt = B * N0 * 3;
  transpose_xyz_kernel<<<dim3((tt + 255) / 256), dim3(256), 0, stream>>>(
      xyz_in, x0, N0, tt);

  const float r1[2] = {0.05f, 0.1f}, r2_[2] = {0.1f, 0.2f},
              r3[2] = {0.2f, 0.4f},  r4[2] = {0.4f, 0.8f};
  const int   ks[2] = {16, 32};

  run_sa(x0,     N0,   x0,  3,   1024, r1,  ks, sa1, l1_xyz, l1p, 96);
  run_sa(l1_xyz, 1024, l1p, 96,  256,  r2_, ks, sa2, l2_xyz, l2p, 256);
  run_sa(l2_xyz, 256,  l2p, 256, 64,   r3,  ks, sa3, l3_xyz, l3p, 512);
  run_sa(l3_xyz, 64,   l3p, 512, 16,   r4,  ks, sa4, l4_xyz, l4p, 1024);

  run_fp(l3_xyz, 64,   l4_xyz, 16,   l3p,     512, l4p, 1024, fp4, 2, l3f);
  run_fp(l2_xyz, 256,  l3_xyz, 64,   l2p,     256, l3f, 256,  fp3, 2, l2f);
  run_fp(l1_xyz, 1024, l2_xyz, 256,  l1p,     96,  l2f, 256,  fp2, 2, l1f);
  const _Float16* headPack =
      run_fp(x0, N0, l1_xyz, 1024, nullptr, 0, l1f, 128, fp1, 3, l0p);

  // ---- head: conv1 + bn_relu, conv2 (+bias) -> d_out ----------------------
  int Mh = B * N0;
  _Float16* other = (headPack == packA) ? packB : packA;
  Lyr c1; c1.W = conv1_W; c1.b = conv1_b; c1.be = bn1_be; c1.g = bn1_g;
  c1.ci = 128; c1.co = 128;
  run_layer(headPack, c1, Mh, true, other, bufY);
  Lyr c2; c2.W = conv2_W; c2.b = conv2_b; c2.be = nullptr; c2.g = nullptr;
  c2.ci = 128; c2.co = 13;
  run_layer(other, c2, Mh, false, nullptr, (float*)d_out);
}
